// ManualEncoderBlock_61753039781968
// MI455X (gfx1250) — matrix operand order repaired and checked by reading, not compiled, not run on hardware
//
#include <hip/hip_runtime.h>
#include <cstddef>

// ---------------- problem dims ----------------
#define BB 4
#define SS 2048
#define DD 1024
#define HH 16
#define DH 64
#define DFF 4096
#define MT (BB*SS)          // 8192 token rows

typedef unsigned short u16;
typedef unsigned int   u32;

typedef __attribute__((ext_vector_type(16))) __bf16 v16bf;
typedef __attribute__((ext_vector_type(8)))  float  v8f;
typedef int v4i_vs __attribute__((vector_size(16)));   // matches builtin param type

union Frag16 { v16bf v; u32 u[8]; };
union U4     { uint4 q;  u16 s[8]; };

// ---------------- CDNA5 async global->LDS copy (guarded) ----------------
#if defined(__has_builtin)
#if __has_builtin(__builtin_amdgcn_global_load_async_to_lds_b128)
#define ASYNC_CP 1
#endif
#endif
#ifndef ASYNC_CP
#define ASYNC_CP 0
#endif

__device__ __forceinline__ void cp16_g2l(const u16* g, u16* l) {
#if ASYNC_CP
  __builtin_amdgcn_global_load_async_to_lds_b128(
      (__attribute__((address_space(1))) v4i_vs*)const_cast<u16*>(g),
      (__attribute__((address_space(3))) v4i_vs*)l, 0, 0);
#else
  *(uint4*)l = *(const uint4*)g;
#endif
}
__device__ __forceinline__ void async_wait() {
#if ASYNC_CP
#if defined(__has_builtin) && __has_builtin(__builtin_amdgcn_s_wait_asynccnt)
  __builtin_amdgcn_s_wait_asynccnt(0);
#else
  asm volatile("s_wait_asynccnt 0x0" ::: "memory");
#endif
#endif
}

__device__ __forceinline__ u16 f2bf(float f) {
  union { float f; u32 u; } c; c.f = f;
  u32 u = c.u;
  u += 0x7FFFu + ((u >> 16) & 1u);          // round-to-nearest-even
  return (u16)(u >> 16);
}
__device__ __forceinline__ float bf2f(u16 h) {
  union { u32 u; float f; } c; c.u = ((u32)h) << 16;
  return c.f;
}

__device__ __forceinline__ v8f wmma_bf16(v16bf a, v16bf b, v8f c) {
  // v_wmma_f32_16x16x32_bf16: (neg_a, A, neg_b, B, c_mod, C, reuse_a, reuse_b)
  return __builtin_amdgcn_wmma_f32_16x16x32_bf16(false, a, false, b, (short)0, c,
                                                 false, false);
}

// ---------------- fp32 -> bf16 convert ----------------
__global__ __launch_bounds__(256) void cvt_f2bf_kernel(const float* __restrict__ src,
                                                       u16* __restrict__ dst,
                                                       long n) {
  long i = (long)blockIdx.x * blockDim.x + threadIdx.x;
  if (i < n) dst[i] = f2bf(src[i]);
}

// ---------------- LayerNorm: fp32 row -> bf16 row ----------------
__global__ __launch_bounds__(256) void ln_kernel(const float* __restrict__ x,
                                                 const float* __restrict__ g,
                                                 const float* __restrict__ b,
                                                 u16* __restrict__ out) {
  __shared__ float red[256];
  const int row = blockIdx.x;
  const float* xr = x + (size_t)row * DD;
  float v[4];
#pragma unroll
  for (int i = 0; i < 4; i++) v[i] = xr[threadIdx.x + i * 256];
  float s = v[0] + v[1] + v[2] + v[3];
  red[threadIdx.x] = s; __syncthreads();
  for (int off = 128; off > 0; off >>= 1) {
    if ((int)threadIdx.x < off) red[threadIdx.x] += red[threadIdx.x + off];
    __syncthreads();
  }
  const float mu = red[0] * (1.0f / DD);
  __syncthreads();
  float s2 = 0.f;
#pragma unroll
  for (int i = 0; i < 4; i++) { float d = v[i] - mu; s2 += d * d; }
  red[threadIdx.x] = s2; __syncthreads();
  for (int off = 128; off > 0; off >>= 1) {
    if ((int)threadIdx.x < off) red[threadIdx.x] += red[threadIdx.x + off];
    __syncthreads();
  }
  const float inv = rsqrtf(red[0] * (1.0f / DD) + 1e-5f);
  u16* orow = out + (size_t)row * DD;
#pragma unroll
  for (int i = 0; i < 4; i++) {
    int c = threadIdx.x + i * 256;
    orow[c] = f2bf((v[i] - mu) * inv * g[c] + b[c]);
  }
}

// ---------------- WMMA GEMM: C[M,N] = A[M,K](bf16) x B[K,N](bf16) + epilogue ----
// EPI 0: none, store bf16
// EPI 1: += bias[n] + resid[m,n], store fp32
// EPI 2: += bias[n], exact GELU, store bf16
#define BM 128
#define BN 64
#define BK 64
#define LDA 72   // padded stride (elements); 144B rows keep 16B-aligned b128 stores
#define LDB 72

template <int EPI, typename OT>
__global__ __launch_bounds__(256) void gemm_bf16_kernel(
    const u16* __restrict__ A, const u16* __restrict__ Bw,
    const float* __restrict__ bias, const float* __restrict__ resid,
    OT* __restrict__ Cout, int M, int N, int K) {
  __shared__ u16 As[BM * LDA];   // [m][k]
  __shared__ u16 Bs[BN * LDB];   // [n][k]  (transposed on staging)

  const int tid  = threadIdx.x;
  const int lane = tid & 31;
  const int wave = tid >> 5;
  const int wm = wave & 3;       // 4 waves along M (32 rows each)
  const int wn = wave >> 2;      // 2 waves along N (32 cols each)
  const int bm0 = blockIdx.y * BM;
  const int bn0 = blockIdx.x * BN;

  v8f zero = {};
  v8f acc[2][2];
#pragma unroll
  for (int i = 0; i < 2; i++)
#pragma unroll
    for (int j = 0; j < 2; j++) acc[i][j] = zero;

  const int arow = tid >> 1;            // 0..127
  const int acol = (tid & 1) * 32;      // 0 / 32
  const int brow = tid >> 2;            // k 0..63
  const int bcol = (tid & 3) * 16;      // n 0,16,32,48

  for (int k0 = 0; k0 < K; k0 += BK) {
    const u16* gA = A + (size_t)(bm0 + arow) * K + k0 + acol;
    const u16* gB = Bw + (size_t)(k0 + brow) * N + bn0 + bcol;
    if (k0 + BK < K) {
      __builtin_prefetch(gA + BK, 0, 0);
      __builtin_prefetch(gB + (size_t)BK * N, 0, 0);
    }
    // stage A tile (128x64): 4 async 16B copies per thread (32 contiguous bf16)
    u16* lA = &As[arow * LDA + acol];
#pragma unroll
    for (int t = 0; t < 4; t++) cp16_g2l(gA + t * 8, lA + t * 8);
    // stage B tile (64x64) transposed to [n][k]
    U4 b0, b1;
    b0.q = ((const uint4*)gB)[0];
    b1.q = ((const uint4*)gB)[1];
#pragma unroll
    for (int e = 0; e < 8; e++) {
      Bs[(bcol + e) * LDB + brow]     = b0.s[e];
      Bs[(bcol + 8 + e) * LDB + brow] = b1.s[e];
    }
    async_wait();
    __syncthreads();

    // two WMMA k-steps per staged tile
#pragma unroll
    for (int c = 0; c < 2; c++) {
      Frag16 af[2], bfr[2];
#pragma unroll
      for (int i = 0; i < 2; i++) {
        const int row = wm * 32 + i * 16 + (lane & 15);
        const int kb  = (lane >> 4) * 8;          // lanes 16-31 start at K=8
#pragma unroll
        for (int v = 0; v < 8; v++) {
          const int kk = c * 32 + kb + (v & 3) * 2 + ((v >> 2) << 4);
          af[i].u[v] = *(const u32*)&As[row * LDA + kk];
        }
      }
#pragma unroll
      for (int j = 0; j < 2; j++) {
        const int n  = wn * 32 + j * 16 + (lane & 15);
        const int kb = (lane & 16) ? 8 : 0;       // lanes 16-31 start at K=8
#pragma unroll
        for (int v = 0; v < 8; v++)
          bfr[j].u[v] = *(const u32*)&Bs[n * LDB + c * 32 + kb + (v & 3) * 2 + ((v >> 2) << 4)];
      }
#pragma unroll
      for (int i = 0; i < 2; i++)
#pragma unroll
        for (int j = 0; j < 2; j++)
          acc[i][j] = wmma_bf16(af[i].v, bfr[j].v, acc[i][j]);
    }
    __syncthreads();
  }

  // epilogue: C layout -> row m = r + 8*(lane>=16), col n = lane&15
#pragma unroll
  for (int i = 0; i < 2; i++)
#pragma unroll
    for (int j = 0; j < 2; j++)
#pragma unroll
      for (int r = 0; r < 8; r++) {
        const int m = bm0 + wm * 32 + i * 16 + r + ((lane >> 4) << 3);
        const int n = bn0 + wn * 32 + j * 16 + (lane & 15);
        float val = acc[i][j][r];
        if constexpr (EPI == 1) val += bias[n] + resid[(size_t)m * N + n];
        if constexpr (EPI == 2) {
          val += bias[n];
          val = 0.5f * val * (1.0f + erff(val * 0.70710678118654752f));
        }
        if constexpr (sizeof(OT) == 2)
          ((u16*)Cout)[(size_t)m * N + n] = f2bf(val);
        else
          ((float*)Cout)[(size_t)m * N + n] = val;
      }
}

// ---------------- RoPE + relayout [B,S,H,DH] -> [B,H,S,DH] (bf16) ------------
__global__ __launch_bounds__(256) void rope_kernel(
    const u16* __restrict__ q, const u16* __restrict__ k,
    const u16* __restrict__ v, const float* __restrict__ freqs,
    u16* __restrict__ qr, u16* __restrict__ kr, u16* __restrict__ vr) {
  const long idx = (long)blockIdx.x * blockDim.x + threadIdx.x;  // B*S*H*32
  const int i = idx & 31;
  const int h = (idx >> 5) & 15;
  const int s = (idx >> 9) & 2047;
  const int b = (int)(idx >> 20);
  const size_t src = (((size_t)(b * SS + s) * HH) + h) * DH + 2 * i;
  const size_t dst = (((size_t)(b * HH + h) * SS) + s) * DH + 2 * i;
  const float c  = freqs[((size_t)s * 32 + i) * 2 + 0];
  const float sn = freqs[((size_t)s * 32 + i) * 2 + 1];
  float q0 = bf2f(q[src]), q1 = bf2f(q[src + 1]);
  float k0 = bf2f(k[src]), k1 = bf2f(k[src + 1]);
  qr[dst]     = f2bf(q0 * c - q1 * sn);
  qr[dst + 1] = f2bf(q0 * sn + q1 * c);
  kr[dst]     = f2bf(k0 * c - k1 * sn);
  kr[dst + 1] = f2bf(k0 * sn + k1 * c);
  vr[dst]     = v[src];
  vr[dst + 1] = v[src + 1];
}

// ---------------- flash attention (bf16 WMMA, fp32 softmax) ------------------
#define QT 128     // q rows per workgroup (8 waves x 16)
#define KT 64      // keys per inner tile
#define LDK 72     // padded LDS stride (144B, 16B-aligned rows)

__global__ __launch_bounds__(256) void attn_kernel(
    const u16* __restrict__ qr, const u16* __restrict__ kr,
    const u16* __restrict__ vr, u16* __restrict__ att) {
  __shared__ u16 Ks[KT * LDK];        // [key][dh]
  __shared__ u16 Vs[DH * LDK];        // [dh][key]   (transposed)
  __shared__ u16 Ps[8 * 16 * LDK];    // per-wave P tile [16][64]

  const int tid  = threadIdx.x;
  const int lane = tid & 31;
  const int wave = tid >> 5;
  const int bh   = blockIdx.y;                 // b*H + h
  const int b    = bh >> 4, h = bh & 15;
  const int q0b  = blockIdx.x * QT;
  const size_t base = (size_t)bh * SS * DH;

  // Q fragments straight from global in A-layout (fixed for whole kernel)
  Frag16 qa[2];
  {
    const int qrow = q0b + wave * 16 + (lane & 15);
    const int kb   = (lane >> 4) * 8;
#pragma unroll
    for (int c = 0; c < 2; c++)
#pragma unroll
      for (int v = 0; v < 8; v++) {
        const int dh = c * 32 + kb + (v & 3) * 2 + ((v >> 2) << 4);
        qa[c].u[v] = *(const u32*)(qr + base + (size_t)qrow * DH + dh);
      }
  }

  v8f zero = {};
  v8f o[4];
#pragma unroll
  for (int j = 0; j < 4; j++) o[j] = zero;
  float m_run[8], l_run[8];
#pragma unroll
  for (int r = 0; r < 8; r++) { m_run[r] = -1e30f; l_run[r] = 0.f; }

  const int srow = tid >> 2;            // 0..63 (key or dh row)
  const int sd0  = (tid & 3) * 16;      // 0,16,32,48

  for (int kb0 = 0; kb0 < SS; kb0 += KT) {
    // ---- stage K via async copy, V transposed manually ----
    {
      const u16* gK = kr + base + (size_t)(kb0 + srow) * DH + sd0;
      u16* lK = &Ks[srow * LDK + sd0];
      cp16_g2l(gK, lK);
      cp16_g2l(gK + 8, lK + 8);
      if (kb0 + KT < SS) {
        __builtin_prefetch(gK + (size_t)KT * DH, 0, 0);
        __builtin_prefetch(vr + base + (size_t)(kb0 + KT + srow) * DH + sd0, 0, 0);
      }
      U4 v0, v1;
      const uint4* pv = (const uint4*)(vr + base + (size_t)(kb0 + srow) * DH + sd0);
      v0.q = pv[0]; v1.q = pv[1];
#pragma unroll
      for (int e = 0; e < 8; e++) {
        Vs[(sd0 + e) * LDK + srow]     = v0.s[e];
        Vs[(sd0 + 8 + e) * LDK + srow] = v1.s[e];
      }
    }
    async_wait();
    __syncthreads();

    // ---- scores: S = Q K^T (16 q-rows x 64 keys per wave) ----
    v8f sc[4];
#pragma unroll
    for (int j = 0; j < 4; j++) sc[j] = zero;
#pragma unroll
    for (int j = 0; j < 4; j++) {
      const int n  = j * 16 + (lane & 15);       // key column
      const int kb = (lane & 16) ? 8 : 0;
#pragma unroll
      for (int c = 0; c < 2; c++) {
        Frag16 bfr;
#pragma unroll
        for (int v = 0; v < 8; v++)
          bfr.u[v] = *(const u32*)&Ks[n * LDK + c * 32 + kb + (v & 3) * 2 + ((v >> 2) << 4)];
        sc[j] = wmma_bf16(qa[c].v, bfr.v, sc[j]);
      }
    }

    // ---- online softmax (rows live per lane-group in C layout) ----
    float mnew[8], alpha[8], rs[8];
#pragma unroll
    for (int r = 0; r < 8; r++) {
#pragma unroll
      for (int j = 0; j < 4; j++) sc[j][r] *= 0.125f;   // 1/sqrt(DH)
      float m = sc[0][r];
#pragma unroll
      for (int j = 1; j < 4; j++) m = fmaxf(m, sc[j][r]);
#pragma unroll
      for (int off = 8; off >= 1; off >>= 1)
        m = fmaxf(m, __shfl_xor(m, off, 32));           // reduce over 16 lanes
      mnew[r] = m;
      const float mt = fmaxf(m_run[r], mnew[r]);
      alpha[r] = __expf(m_run[r] - mt);
      m_run[r] = mt;
      float s = 0.f;
#pragma unroll
      for (int j = 0; j < 4; j++) {
        float p = __expf(sc[j][r] - mt);
        sc[j][r] = p;
        s += p;
      }
#pragma unroll
      for (int off = 8; off >= 1; off >>= 1)
        s += __shfl_xor(s, off, 32);
      rs[r] = s;
      l_run[r] = l_run[r] * alpha[r] + rs[r];
#pragma unroll
      for (int j = 0; j < 4; j++) o[j][r] *= alpha[r];
    }

    // ---- transpose P through LDS into A-layout ----
    const int pbase = wave * 16;
#pragma unroll
    for (int j = 0; j < 4; j++)
#pragma unroll
      for (int r = 0; r < 8; r++) {
        const int m = pbase + r + ((lane >> 4) << 3);
        Ps[m * LDK + j * 16 + (lane & 15)] = f2bf(sc[j][r]);
      }
    __syncthreads();

    // ---- O += P V ----
#pragma unroll
    for (int c = 0; c < 2; c++) {
      Frag16 pf;
      const int m  = pbase + (lane & 15);
      const int kb = (lane >> 4) * 8;
#pragma unroll
      for (int v = 0; v < 8; v++) {
        const int key = c * 32 + kb + (v & 3) * 2 + ((v >> 2) << 4);
        pf.u[v] = *(const u32*)&Ps[m * LDK + key];
      }
#pragma unroll
      for (int j = 0; j < 4; j++) {
        Frag16 vf;
        const int n   = j * 16 + (lane & 15);     // dh column
        const int kb2 = (lane & 16) ? 8 : 0;
#pragma unroll
        for (int v = 0; v < 8; v++)
          vf.u[v] = *(const u32*)&Vs[n * LDK + c * 32 + kb2 + (v & 3) * 2 + ((v >> 2) << 4)];
        o[j] = wmma_bf16(pf.v, vf.v, o[j]);
      }
    }
    __syncthreads();
  }

  // ---- finalize: divide by l, store [B,S,H,DH] bf16 ----
#pragma unroll
  for (int j = 0; j < 4; j++)
#pragma unroll
    for (int r = 0; r < 8; r++) {
      const int m  = q0b + wave * 16 + r + ((lane >> 4) << 3);
      const int dh = j * 16 + (lane & 15);
      const float val = o[j][r] / l_run[r];
      att[(((size_t)(b * SS + m) * HH) + h) * DH + dh] = f2bf(val);
    }
}

// ---------------- host orchestration ----------------
extern "C" void kernel_launch(void* const* d_in, const int* in_sizes, int n_in,
                              void* d_out, int out_size, void* d_ws, size_t ws_size,
                              hipStream_t stream) {
  const float* x      = (const float*)d_in[0];
  const float* freqs  = (const float*)d_in[1];
  const float* wq     = (const float*)d_in[2];
  const float* wk     = (const float*)d_in[3];
  const float* wv     = (const float*)d_in[4];
  const float* wproj  = (const float*)d_in[5];
  const float* bproj  = (const float*)d_in[6];
  const float* wff1   = (const float*)d_in[7];
  const float* bff1   = (const float*)d_in[8];
  const float* wff2   = (const float*)d_in[9];
  const float* bff2   = (const float*)d_in[10];
  const float* ln1g   = (const float*)d_in[11];
  const float* ln1b   = (const float*)d_in[12];
  const float* ln2g   = (const float*)d_in[13];
  const float* ln2b   = (const float*)d_in[14];
  float* out = (float*)d_out;

  char* ws = (char*)d_ws;
  size_t off = 0;
  auto alloc = [&](size_t bytes) -> void* {
    void* p = ws + off;
    off = (off + bytes + 255) & ~(size_t)255;
    return p;
  };
  u16* wq_bf   = (u16*)alloc((size_t)DD * DD * 2);
  u16* wk_bf   = (u16*)alloc((size_t)DD * DD * 2);
  u16* wv_bf   = (u16*)alloc((size_t)DD * DD * 2);
  u16* wp_bf   = (u16*)alloc((size_t)DD * DD * 2);
  u16* wff1_bf = (u16*)alloc((size_t)DD * DFF * 2);
  u16* wff2_bf = (u16*)alloc((size_t)DFF * DD * 2);
  u16* h_bf    = (u16*)alloc((size_t)MT * DD * 2);
  u16* q_bf    = (u16*)alloc((size_t)MT * DD * 2);
  u16* k_bf    = (u16*)alloc((size_t)MT * DD * 2);
  u16* v_bf    = (u16*)alloc((size_t)MT * DD * 2);
  u16* q_r     = (u16*)alloc((size_t)MT * DD * 2);
  u16* k_r     = (u16*)alloc((size_t)MT * DD * 2);
  u16* v_r     = (u16*)alloc((size_t)MT * DD * 2);
  u16* att_bf  = (u16*)alloc((size_t)MT * DD * 2);
  float* x1    = (float*)alloc((size_t)MT * DD * 4);
  u16* h2_bf   = (u16*)alloc((size_t)MT * DD * 2);
  u16* ff_bf   = (u16*)alloc((size_t)MT * DFF * 2);

  const dim3 blk(256);
  // weights -> bf16
  {
    long n1 = (long)DD * DD, n2 = (long)DD * DFF;
    cvt_f2bf_kernel<<<dim3((n1 + 255) / 256), blk, 0, stream>>>(wq, wq_bf, n1);
    cvt_f2bf_kernel<<<dim3((n1 + 255) / 256), blk, 0, stream>>>(wk, wk_bf, n1);
    cvt_f2bf_kernel<<<dim3((n1 + 255) / 256), blk, 0, stream>>>(wv, wv_bf, n1);
    cvt_f2bf_kernel<<<dim3((n1 + 255) / 256), blk, 0, stream>>>(wproj, wp_bf, n1);
    cvt_f2bf_kernel<<<dim3((n2 + 255) / 256), blk, 0, stream>>>(wff1, wff1_bf, n2);
    cvt_f2bf_kernel<<<dim3((n2 + 255) / 256), blk, 0, stream>>>(wff2, wff2_bf, n2);
  }
  // LN1
  ln_kernel<<<dim3(MT), blk, 0, stream>>>(x, ln1g, ln1b, h_bf);
  // QKV GEMMs
  {
    dim3 g(DD / BN, MT / BM);
    gemm_bf16_kernel<0, u16><<<g, blk, 0, stream>>>(h_bf, wq_bf, nullptr, nullptr, q_bf, MT, DD, DD);
    gemm_bf16_kernel<0, u16><<<g, blk, 0, stream>>>(h_bf, wk_bf, nullptr, nullptr, k_bf, MT, DD, DD);
    gemm_bf16_kernel<0, u16><<<g, blk, 0, stream>>>(h_bf, wv_bf, nullptr, nullptr, v_bf, MT, DD, DD);
  }
  // RoPE + relayout
  {
    long n = (long)BB * SS * HH * 32;
    rope_kernel<<<dim3(n / 256), blk, 0, stream>>>(q_bf, k_bf, v_bf, freqs, q_r, k_r, v_r);
  }
  // attention
  attn_kernel<<<dim3(SS / QT, BB * HH), blk, 0, stream>>>(q_r, k_r, v_r, att_bf);
  // proj + bias + residual(x) -> x1 (fp32)
  gemm_bf16_kernel<1, float><<<dim3(DD / BN, MT / BM), blk, 0, stream>>>(
      att_bf, wp_bf, bproj, x, x1, MT, DD, DD);
  // LN2
  ln_kernel<<<dim3(MT), blk, 0, stream>>>(x1, ln2g, ln2b, h2_bf);
  // FF1 + bias + GELU -> bf16
  gemm_bf16_kernel<2, u16><<<dim3(DFF / BN, MT / BM), blk, 0, stream>>>(
      h2_bf, wff1_bf, bff1, nullptr, ff_bf, MT, DD, DFF);
  // FF2 + bias + residual(x1) -> out (fp32)
  gemm_bf16_kernel<1, float><<<dim3(DD / BN, MT / BM), blk, 0, stream>>>(
      ff_bf, wff2_bf, bff2, x1, out, MT, DFF, DD);
}
